// MultiHeadAttention_85718957293752
// MI455X (gfx1250) — matrix operand order repaired and checked by reading, not compiled, not run on hardware
//
#include <hip/hip_runtime.h>
#include <hip/hip_bf16.h>

typedef __attribute__((ext_vector_type(16))) _Float16 v16h;
typedef __attribute__((ext_vector_type(8)))  _Float16 v8h;
typedef __attribute__((ext_vector_type(8)))  float    v8f;
typedef __attribute__((ext_vector_type(4)))  float    v4f;

namespace {
constexpr int kB   = 8;
constexpr int kL   = 4096;   // power of two -> & (kL-1) for circular wrap
constexpr int kD   = 512;    // D == H*DK == H*DV
constexpr int kTop = 8;      // int(log(4096)) == 8
constexpr int kLdsPitch = 48; // halves per LDS row (32 data + 16 pad, 96B rows)

constexpr size_t kActBytes = (size_t)kB * kL * kD * sizeof(_Float16); // 32 MB
constexpr size_t kWtBytes  = (size_t)kD * kD * sizeof(_Float16);      // 512 KB
constexpr size_t OFF_QH   = 0;
constexpr size_t OFF_KH   = kActBytes;
constexpr size_t OFF_VH   = 2 * kActBytes;
constexpr size_t OFF_CTX  = 3 * kActBytes;
constexpr size_t OFF_WQT  = 4 * kActBytes;
constexpr size_t OFF_WKT  = OFF_WQT + kWtBytes;
constexpr size_t OFF_WVT  = OFF_WQT + 2 * kWtBytes;
constexpr size_t OFF_WFCT = OFF_WQT + 3 * kWtBytes;
constexpr size_t OFF_MV   = OFF_WQT + 4 * kWtBytes;                   // B*L f32
constexpr size_t OFF_IDX  = OFF_MV + (size_t)kB * kL * sizeof(float); // 8 ints
constexpr size_t OFF_TC   = OFF_IDX + 256;                            // B*8 f32
} // namespace

// ---------------- WMMA fragment helpers (V_WMMA_F32_16X16X32_F16) ----------
// A 16x32 f16 layout (ISA 7.12.2): lane<16 -> M=lane, halves0..7 = K0..7,
// halves8..15 = K16..23; lane>=16 -> same M, K ranges +8.
__device__ __forceinline__ v16h load_a_f16(const _Float16* __restrict__ base,
                                           int ld, int row0, int k0, int lane) {
  const int m  = lane & 15;
  const int kb = (lane & 16) ? 8 : 0;
  const _Float16* r = base + (size_t)(row0 + m) * ld + k0 + kb;
  v8h lo = *(const v8h*)r;
  v8h hi = *(const v8h*)(r + 16);
  v16h a;
#pragma unroll
  for (int i = 0; i < 8; ++i) { a[i] = lo[i]; a[i + 8] = hi[i]; }
  return a;
}

__device__ __forceinline__ v16h load_a_f32(const float* __restrict__ base,
                                           int ld, int row0, int k0, int lane) {
  const int m  = lane & 15;
  const int kb = (lane & 16) ? 8 : 0;
  const float* r = base + (size_t)(row0 + m) * ld + k0 + kb;
  v4f a0 = *(const v4f*)(r);
  v4f a1 = *(const v4f*)(r + 4);
  v4f b0 = *(const v4f*)(r + 16);
  v4f b1 = *(const v4f*)(r + 20);
  v16h a;
#pragma unroll
  for (int i = 0; i < 4; ++i) {
    a[i]      = (_Float16)a0[i];
    a[i + 4]  = (_Float16)a1[i];
    a[i + 8]  = (_Float16)b0[i];
    a[i + 12] = (_Float16)b1[i];
  }
  return a;
}

// B 32x16 f16: lane holds column N=lane&15, halves0..7 = K kb..kb+7,
// halves8..15 = K kb+16..kb+23, kb = (lane&16)?8:0. Source is "N-major" (contiguous K), i.e. k-rows for
// the NT correlation GEMM, or pre-transposed weights.
__device__ __forceinline__ v16h load_b_nt(const _Float16* __restrict__ base,
                                          int ld, int n0, int k0, int lane) {
  const int n  = lane & 15;
  const int kb = (lane & 16) ? 8 : 0;
  const _Float16* r = base + (size_t)(n0 + n) * ld + k0 + kb;
  v8h lo = *(const v8h*)r;
  v8h hi = *(const v8h*)(r + 16);
  v16h b;
#pragma unroll
  for (int i = 0; i < 8; ++i) { b[i] = lo[i]; b[i + 8] = hi[i]; }
  return b;
}

// LDS variants: slab holds K=32 halves per row, row pitch kLdsPitch halves.
__device__ __forceinline__ v16h lds_load_a(const _Float16* base, int row0, int lane) {
  const int m  = lane & 15;
  const int kb = (lane & 16) ? 8 : 0;
  const _Float16* r = base + (row0 + m) * kLdsPitch + kb;
  v8h lo = *(const v8h*)r;
  v8h hi = *(const v8h*)(r + 16);
  v16h a;
#pragma unroll
  for (int i = 0; i < 8; ++i) { a[i] = lo[i]; a[i + 8] = hi[i]; }
  return a;
}

__device__ __forceinline__ v16h lds_load_b(const _Float16* base, int n0, int lane) {
  const int n  = lane & 15;
  const int kb = (lane & 16) ? 8 : 0;
  const _Float16* r = base + (n0 + n) * kLdsPitch + kb;
  v8h lo = *(const v8h*)r;
  v8h hi = *(const v8h*)(r + 16);
  v16h b;
#pragma unroll
  for (int i = 0; i < 8; ++i) { b[i] = lo[i]; b[i + 8] = hi[i]; }
  return b;
}

__device__ __forceinline__ v8f wmma_f16(v16h a, v16h b, v8f c) {
  return __builtin_amdgcn_wmma_f32_16x16x32_f16(false, a, false, b, (short)0, c,
                                                false, false);
}

__device__ __forceinline__ v8f vzero8() {
  return (v8f){0.f, 0.f, 0.f, 0.f, 0.f, 0.f, 0.f, 0.f};
}

// ---------------- gfx1250 async copy global -> LDS (ASYNCcnt-tracked) ------
// GVS mode: mem = SADDR(u64) + VADDR(u32); per-lane 16B into LDS[VDST vgpr].
__device__ __forceinline__ void async_b128(unsigned loff, unsigned goff,
                                           const _Float16* sbase) {
  asm volatile("global_load_async_to_lds_b128 %0, %1, %2"
               :
               : "v"(loff), "v"(goff), "s"(sbase)
               : "memory");
}
__device__ __forceinline__ void wait_async_le4() {
  asm volatile("s_wait_asynccnt 4" ::: "memory");
}
__device__ __forceinline__ void wait_async_0() {
  asm volatile("s_wait_asynccnt 0" ::: "memory");
}

// ---------------- small prep kernels ---------------------------------------
__global__ __launch_bounds__(256) void zero_f32_kernel(float* __restrict__ p, int n) {
  int i = blockIdx.x * 256 + threadIdx.x;
  if (i < n) p[i] = 0.f;
}

// Wt[n*K + k] = (f16) W[k*N + n]
__global__ __launch_bounds__(256) void wtr_kernel(const float* __restrict__ W,
                                                  _Float16* __restrict__ Wt) {
  int g = blockIdx.x * 256 + threadIdx.x; // over kD*kD
  int n = g & (kD - 1);
  int k = g >> 9; // /512
  Wt[(size_t)n * kD + k] = (_Float16)W[(size_t)k * kD + n];
}

// ---------------- projection GEMM: f32 A x f16 Bt -> f16 out ---------------
__global__ __launch_bounds__(256, 2)
void proj_gemm_kernel(const float* __restrict__ A, const _Float16* __restrict__ Bt,
                      _Float16* __restrict__ Cout) {
  const int lane = threadIdx.x & 31;
  const int w    = threadIdx.x >> 5;
  const int wi   = w & 3;
  const int wj   = w >> 2;
  const int i0   = blockIdx.x * 128 + wi * 32;
  const int j0   = blockIdx.y * 128 + wj * 64;
  v8f c[2][4];
#pragma unroll
  for (int mi = 0; mi < 2; ++mi)
#pragma unroll
    for (int nj = 0; nj < 4; ++nj) c[mi][nj] = vzero8();

  for (int kk = 0; kk < kD; kk += 32) {
    v16h a[2], b[4];
#pragma unroll
    for (int mi = 0; mi < 2; ++mi) a[mi] = load_a_f32(A, kD, i0 + 16 * mi, kk, lane);
#pragma unroll
    for (int nj = 0; nj < 4; ++nj) b[nj] = load_b_nt(Bt, kD, j0 + 16 * nj, kk, lane);
#pragma unroll
    for (int mi = 0; mi < 2; ++mi)
#pragma unroll
      for (int nj = 0; nj < 4; ++nj) c[mi][nj] = wmma_f16(a[mi], b[nj], c[mi][nj]);
  }

  const int mo = (lane & 16) ? 8 : 0;
  const int n  = lane & 15;
#pragma unroll
  for (int mi = 0; mi < 2; ++mi)
#pragma unroll
    for (int nj = 0; nj < 4; ++nj)
#pragma unroll
      for (int r = 0; r < 8; ++r) {
        int row = i0 + 16 * mi + r + mo;
        int col = j0 + 16 * nj + n;
        Cout[(size_t)row * kD + col] = (_Float16)c[mi][nj][r];
      }
}

// ---------------- correlation GEMM, async-LDS double-buffered --------------
// mean_value[b,tau] += sum over tile of q[b,i,:]·k[b,j,:] where tau=(i-j)%L.
// K-slabs (128 rows x 32 halves) of q and k staged into LDS with
// global_load_async_to_lds_b128; consumed by 8 WMMAs per wave per slab.
__global__ __launch_bounds__(256, 2)
void corr_gemm_kernel(const _Float16* __restrict__ qh, const _Float16* __restrict__ kh,
                      float* __restrict__ mv) {
  __shared__ _Float16 As[2][128 * kLdsPitch];
  __shared__ _Float16 Bs[2][128 * kLdsPitch];
  __shared__ float    hist[255];

  const int b    = blockIdx.z;
  const _Float16* Aq = qh + (size_t)b * kL * kD;
  const _Float16* Bk = kh + (size_t)b * kL * kD;
  const int tid  = threadIdx.x;
  const int lane = tid & 31;
  const int w    = tid >> 5;
  const int wi   = w & 3;
  const int wj   = w >> 2;
  const int di0  = blockIdx.x * 128;
  const int dj0  = blockIdx.y * 128;

  for (int t = tid; t < 255; t += 256) hist[t] = 0.f;

  // Async-copy task split: 128 rows x 4 x 16B segments = 512 tasks for A and
  // for B; each of the 256 threads issues 2 A-loads + 2 B-loads per slab.
  const int rr = tid >> 1;
  const int s0 = (tid & 1) * 2;
  auto issue_slab = [&](int buf, int kk) {
    unsigned lA = (unsigned)(size_t)&As[buf][rr * kLdsPitch + s0 * 8];
    unsigned lB = (unsigned)(size_t)&Bs[buf][rr * kLdsPitch + s0 * 8];
    unsigned gA = (unsigned)((((di0 + rr) * kD) + kk + s0 * 8) * 2);
    unsigned gB = (unsigned)((((dj0 + rr) * kD) + kk + s0 * 8) * 2);
    async_b128(lA, gA, Aq);
    async_b128(lA + 16u, gA + 16u, Aq);
    async_b128(lB, gB, Bk);
    async_b128(lB + 16u, gB + 16u, Bk);
  };

  v8f c[2][4];
#pragma unroll
  for (int mi = 0; mi < 2; ++mi)
#pragma unroll
    for (int nj = 0; nj < 4; ++nj) c[mi][nj] = vzero8();

  issue_slab(0, 0); // prologue: fill buffer 0

  for (int kk = 0; kk < kD; kk += 32) {
    const int cur = (kk >> 5) & 1;
    if (kk + 32 < kD) {
      __syncthreads();              // everyone done reading buffer cur^1
      issue_slab(cur ^ 1, kk + 32); // prefetch next slab
      wait_async_le4();             // own loads of slab 'cur' complete (FIFO)
    } else {
      wait_async_0();
    }
    __syncthreads();                // all waves' slab 'cur' resident in LDS

    v16h a[2], bfr[4];
#pragma unroll
    for (int mi = 0; mi < 2; ++mi)
      a[mi] = lds_load_a(&As[cur][0], wi * 32 + mi * 16, lane);
#pragma unroll
    for (int nj = 0; nj < 4; ++nj)
      bfr[nj] = lds_load_b(&Bs[cur][0], wj * 64 + nj * 16, lane);
#pragma unroll
    for (int mi = 0; mi < 2; ++mi)
#pragma unroll
      for (int nj = 0; nj < 4; ++nj) c[mi][nj] = wmma_f16(a[mi], bfr[nj], c[mi][nj]);
  }

  const int mo = (lane & 16) ? 8 : 0;
  const int n  = lane & 15;
#pragma unroll
  for (int mi = 0; mi < 2; ++mi)
#pragma unroll
    for (int nj = 0; nj < 4; ++nj)
#pragma unroll
      for (int r = 0; r < 8; ++r) {
        int irel = wi * 32 + mi * 16 + r + mo; // i - di0, in [0,128)
        int jrel = wj * 64 + nj * 16 + n;      // j - dj0, in [0,128)
        atomicAdd(&hist[irel - jrel + 127], c[mi][nj][r]); // ds_add_f32
      }
  __syncthreads();

  for (int t = tid; t < 255; t += 256) {
    int tau = (di0 - dj0 + t - 127) & (kL - 1);
    atomicAdd(&mv[(size_t)b * kL + tau], hist[t]); // global_atomic_add_f32
  }
}

// ---------------- top-k lags + per-batch softmax ---------------------------
__global__ __launch_bounds__(256)
void topk_softmax_kernel(const float* __restrict__ mv, int* __restrict__ idx,
                         float* __restrict__ tc) {
  __shared__ float sv[kL];
  __shared__ float rv[256];
  __shared__ int   ri[256];
  __shared__ int   bi[kTop];
  const int tid = threadIdx.x;

  for (int t = tid; t < kL; t += 256) {
    float s = 0.f;
    for (int b = 0; b < kB; ++b) s += mv[(size_t)b * kL + t];
    sv[t] = s;
  }
  __syncthreads();

  for (int it = 0; it < kTop; ++it) {
    float best = -3.4e38f;
    int   bj   = 0;
    for (int t = tid; t < kL; t += 256)
      if (sv[t] > best) { best = sv[t]; bj = t; }
    rv[tid] = best;
    ri[tid] = bj;
    __syncthreads();
    if (tid == 0) {
      float bb = rv[0];
      int   jj = ri[0];
      for (int u = 1; u < 256; ++u)
        if (rv[u] > bb) { bb = rv[u]; jj = ri[u]; }
      bi[it] = jj;
      sv[jj] = -3.4e38f;
    }
    __syncthreads();
  }

  if (tid == 0)
    for (int i = 0; i < kTop; ++i) idx[i] = bi[i];

  if (tid < kB) { // per-batch softmax over the selected lags (mean scale 1/512)
    const int b = tid;
    float wv[kTop], mx = -3.4e38f;
    for (int i = 0; i < kTop; ++i) {
      wv[i] = mv[(size_t)b * kL + bi[i]] * (1.0f / (float)kD);
      mx    = fmaxf(mx, wv[i]);
    }
    float s = 0.f;
    for (int i = 0; i < kTop; ++i) { wv[i] = __expf(wv[i] - mx); s += wv[i]; }
    for (int i = 0; i < kTop; ++i) tc[b * kTop + i] = wv[i] / s;
  }
}

// ---------------- delay aggregation: ctx[b,t,c] = sum_i w_i * v[b,(t+d_i)%L,c]
__global__ __launch_bounds__(256)
void delay_agg_kernel(const _Float16* __restrict__ vh, const int* __restrict__ idx,
                      const float* __restrict__ tc, _Float16* __restrict__ ctx) {
  const size_t gid = (size_t)blockIdx.x * 256 + threadIdx.x; // one per 8 channels
  const int cb = (int)(gid & 63);          // 64 groups of 8 channels
  const size_t tmp = gid >> 6;
  const int t = (int)(tmp & (kL - 1));
  const int b = (int)(tmp >> 12);
  int   d[kTop];
  float w[kTop];
#pragma unroll
  for (int i = 0; i < kTop; ++i) { d[i] = idx[i]; w[i] = tc[b * kTop + i]; }
  const _Float16* vb = vh + (size_t)b * kL * kD;
  float acc[8] = {0.f, 0.f, 0.f, 0.f, 0.f, 0.f, 0.f, 0.f};
#pragma unroll
  for (int i = 0; i < kTop; ++i) {
    int tt = (t + d[i]) & (kL - 1);
    v8h x  = *(const v8h*)(vb + (size_t)tt * kD + cb * 8);
#pragma unroll
    for (int u = 0; u < 8; ++u) acc[u] += w[i] * (float)x[u];
  }
  v8h o;
#pragma unroll
  for (int u = 0; u < 8; ++u) o[u] = (_Float16)acc[u];
  *(v8h*)(ctx + ((size_t)b * kL + t) * kD + cb * 8) = o;
}

// ---------------- final GEMM: f16 ctx x f16 Wfc^T -> f32 out ---------------
__global__ __launch_bounds__(256, 2)
void final_gemm_kernel(const _Float16* __restrict__ A, const _Float16* __restrict__ Bt,
                       float* __restrict__ Cout) {
  const int lane = threadIdx.x & 31;
  const int w    = threadIdx.x >> 5;
  const int wi   = w & 3;
  const int wj   = w >> 2;
  const int i0   = blockIdx.x * 128 + wi * 32;
  const int j0   = blockIdx.y * 128 + wj * 64;
  v8f c[2][4];
#pragma unroll
  for (int mi = 0; mi < 2; ++mi)
#pragma unroll
    for (int nj = 0; nj < 4; ++nj) c[mi][nj] = vzero8();

  for (int kk = 0; kk < kD; kk += 32) {
    v16h a[2], b[4];
#pragma unroll
    for (int mi = 0; mi < 2; ++mi) a[mi] = load_a_f16(A, kD, i0 + 16 * mi, kk, lane);
#pragma unroll
    for (int nj = 0; nj < 4; ++nj) b[nj] = load_b_nt(Bt, kD, j0 + 16 * nj, kk, lane);
#pragma unroll
    for (int mi = 0; mi < 2; ++mi)
#pragma unroll
      for (int nj = 0; nj < 4; ++nj) c[mi][nj] = wmma_f16(a[mi], b[nj], c[mi][nj]);
  }

  const int mo = (lane & 16) ? 8 : 0;
  const int n  = lane & 15;
#pragma unroll
  for (int mi = 0; mi < 2; ++mi)
#pragma unroll
    for (int nj = 0; nj < 4; ++nj)
#pragma unroll
      for (int r = 0; r < 8; ++r) {
        int row = i0 + 16 * mi + r + mo;
        int col = j0 + 16 * nj + n;
        Cout[(size_t)row * kD + col] = c[mi][nj][r];
      }
}

// ---------------- host-side orchestration ----------------------------------
extern "C" void kernel_launch(void* const* d_in, const int* in_sizes, int n_in,
                              void* d_out, int out_size, void* d_ws, size_t ws_size,
                              hipStream_t stream) {
  const float* Q   = (const float*)d_in[0];
  const float* K   = (const float*)d_in[1];
  const float* V   = (const float*)d_in[2];
  // d_in[3] = attn_mask (unused)
  const float* WQ  = (const float*)d_in[4];
  const float* WK  = (const float*)d_in[5];
  const float* WV  = (const float*)d_in[6];
  const float* Wfc = (const float*)d_in[7];
  float* out = (float*)d_out;

  char* ws = (char*)d_ws;
  _Float16* qh   = (_Float16*)(ws + OFF_QH);
  _Float16* kh   = (_Float16*)(ws + OFF_KH);
  _Float16* vh   = (_Float16*)(ws + OFF_VH);
  _Float16* ctx  = (_Float16*)(ws + OFF_CTX);
  _Float16* wqt  = (_Float16*)(ws + OFF_WQT);
  _Float16* wkt  = (_Float16*)(ws + OFF_WKT);
  _Float16* wvt  = (_Float16*)(ws + OFF_WVT);
  _Float16* wfct = (_Float16*)(ws + OFF_WFCT);
  float*    mv   = (float*)(ws + OFF_MV);
  int*      idx  = (int*)(ws + OFF_IDX);
  float*    tc   = (float*)(ws + OFF_TC);

  const int nMV = kB * kL;
  zero_f32_kernel<<<(nMV + 255) / 256, 256, 0, stream>>>(mv, nMV);

  const int nWT = kD * kD;
  wtr_kernel<<<nWT / 256, 256, 0, stream>>>(WQ, wqt);
  wtr_kernel<<<nWT / 256, 256, 0, stream>>>(WK, wkt);
  wtr_kernel<<<nWT / 256, 256, 0, stream>>>(WV, wvt);
  wtr_kernel<<<nWT / 256, 256, 0, stream>>>(Wfc, wfct);

  dim3 gg((kB * kL) / 128, kD / 128); // 256 x 4
  proj_gemm_kernel<<<gg, 256, 0, stream>>>(Q, wqt, qh);
  proj_gemm_kernel<<<gg, 256, 0, stream>>>(K, wkt, kh);
  proj_gemm_kernel<<<gg, 256, 0, stream>>>(V, wvt, vh);

  dim3 gc(kL / 128, kL / 128, kB); // 32 x 32 x 8
  corr_gemm_kernel<<<gc, 256, 0, stream>>>(qh, kh, mv);

  topk_softmax_kernel<<<1, 256, 0, stream>>>(mv, idx, tc);

  delay_agg_kernel<<<(kB * kL * (kD / 8)) / 256, 256, 0, stream>>>(vh, idx, tc, ctx);

  final_gemm_kernel<<<gg, 256, 0, stream>>>(ctx, wfct, out);
}
